// Model_18253611008394
// MI455X (gfx1250) — matrix operand order repaired and checked by reading, not compiled, not run on hardware
//
#include <hip/hip_runtime.h>
#include <cstddef>
#include <cstdint>

// ============================================================================
// Point Transformer forward for MI455X (gfx1250, wave32, WMMA).
// All dense linears run through a f16-WMMA GEMM (f32 accumulate) with LDS
// fragment staging; everything else is bandwidth-bound elementwise work.
// ============================================================================

typedef __attribute__((ext_vector_type(16))) _Float16 v16h;
typedef __attribute__((ext_vector_type(8)))  float    v8f;

// explicit global-address-space pointer so staging loads stay global_load_*
// (pointer arrays + loop PHIs otherwise defeat addrspace inference -> flat_load)
typedef const float __attribute__((address_space(1)))* gcf;

// ---------------------------------------------------------------------------
// GEMM: C[M,N] = act(A[M,K] @ W[K,N] + bias), f32 in/out, f16 WMMA compute.
// Block tile 128x64, 256 threads = 8 waves laid out 4(M) x 2(N); each wave
// owns a 32x32 output tile = 4 x v_wmma_f32_16x16x32_f16 per K-step
// (2 A-frags x 2 B-frags, register reuse).
// LDS holds A/B tiles pre-swizzled into WMMA fragment layout:
//   A (16x32 f16): lane = row + ((k>>3)&1)*16 ; half = (k&7) + ((k>>4)&1)*8
//   B (32x16 f16): lane = n + ((k>>3)&1)*16 ; half = (k&7) + ((k>>4)&1)*8
// so each lane reads its v16h as one contiguous 32B chunk (ds_load_b128 x2).
// Row/col clamps + LDS store addresses are loop-invariant and hoisted; the
// main K-loop covers full 32-wide tiles with pure pointer-increment loads
// (one clause, one wait), a single tail iteration handles ragged K.
// ---------------------------------------------------------------------------
__global__ __launch_bounds__(256)
void gemm_f16wmma(const float* __restrict__ A, const float* __restrict__ W,
                  const float* __restrict__ bias, float* __restrict__ C,
                  int M, int K, int N, int relu)
{
    __shared__ _Float16 As[4096] __attribute__((aligned(32))); // 8 groups * 512
    __shared__ _Float16 Bs[2048] __attribute__((aligned(32))); // 4 ntiles * 512

    const int tid  = threadIdx.x;
    const int lane = tid & 31;
    const int wave = tid >> 5;
    const int wm   = wave & 3;   // M quarter (32 rows each)
    const int wn   = wave >> 2;  // N half   (32 cols each)
    const int row0 = blockIdx.y * 128;
    const int col0 = blockIdx.x * 64;

    v8f acc00 = {}, acc01 = {}, acc10 = {}, acc11 = {};

    // ---- loop-invariant staging state (hoisted out of the K loop) --------
    // A: 8 pairs/thread; row clamp + LDS target fixed per i.
    gcf aPtr[8]; bool aOk[8]; _Float16* aLds[8];
    size_t aOff0 = 0;
#pragma unroll
    for (int i = 0; i < 8; ++i) {
        int pe = tid + (i << 8);
        int r  = pe >> 4;              // 0..127
        int k  = (pe & 15) << 1;       // 0..30 even
        int gr = row0 + r;
        aOk[i] = gr < M;
        int cr = aOk[i] ? gr : (M - 1);
        size_t off = (size_t)cr * K + k;
        if (i == 0) aOff0 = off;
        aPtr[i] = (gcf)(A + off);
        int grp = r >> 4, rr = r & 15;
        int l = rr + (((k >> 3) & 1) << 4);
        int h = (k & 7) + (((k >> 4) & 1) << 3);   // even
        aLds[i] = As + grp * 512 + l * 16 + h;
    }
    // B: 4 pairs/thread; col clamp + LDS target fixed per i.
    gcf bPtr0[4]; gcf bPtr1[4];
    bool bOk0[4], bOk1[4]; _Float16* bLds[4];
    size_t bOff0 = 0;
#pragma unroll
    for (int i = 0; i < 4; ++i) {
        int pe = tid + (i << 8);
        int kk = pe >> 5;              // 0..31
        int n  = (pe & 31) << 1;       // 0..62 even
        int gn = col0 + n;
        bOk0[i] = gn < N;
        bOk1[i] = (gn + 1) < N;
        int cn0 = bOk0[i] ? gn : (N - 1);
        int cn1 = bOk1[i] ? (gn + 1) : (N - 1);
        if (i == 0) bOff0 = (size_t)kk * N + cn0;
        bPtr0[i] = (gcf)(W + (size_t)kk * N + cn0);
        bPtr1[i] = (gcf)(W + (size_t)kk * N + cn1);
        bLds[i] = Bs + (n >> 4) * 512 + ((n & 15) + (((kk >> 3) & 1) << 4)) * 16 + (kk & 7) + (((kk >> 4) & 1) << 3);
    }
    const size_t bStep = (size_t)32 * N;
    const int Kfull = K & ~31;

    // ======================= main loop: full K tiles =======================
    for (int k0 = 0; k0 < Kfull; k0 += 32) {
        float va0[8], va1[8];
#pragma unroll
        for (int i = 0; i < 8; ++i) {
            va0[i] = aPtr[i][0];
            va1[i] = aPtr[i][1];
            aPtr[i] += 32;
        }
        float vb0[4], vb1[4];
#pragma unroll
        for (int i = 0; i < 4; ++i) {
            vb0[i] = bPtr0[i][0];
            vb1[i] = bPtr1[i][0];
            bPtr0[i] += bStep;
            bPtr1[i] += bStep;
        }
        if (k0 + 32 < K) {             // speculative prefetch of next tile
            __builtin_prefetch(A + aOff0 + k0 + 32, 0, 1);
            __builtin_prefetch(W + bOff0 + (size_t)(k0 + 32) * N, 0, 1);
        }

        __syncthreads();   // previous iteration's fragment reads complete
#pragma unroll
        for (int i = 0; i < 8; ++i) {
            union { _Float16 h[2]; unsigned u; } pk;
            pk.h[0] = (_Float16)(aOk[i] ? va0[i] : 0.0f);
            pk.h[1] = (_Float16)(aOk[i] ? va1[i] : 0.0f);
            *(unsigned*)aLds[i] = pk.u;
        }
#pragma unroll
        for (int i = 0; i < 4; ++i) {
            union { _Float16 h[2]; unsigned u; } pk;
            pk.h[0] = (_Float16)(bOk0[i] ? vb0[i] : 0.0f);
            pk.h[1] = (_Float16)(bOk1[i] ? vb1[i] : 0.0f);
            bLds[i][0] = pk.h[0]; bLds[i][16] = pk.h[1];
        }
        __syncthreads();

        v16h a0 = *(const v16h*)(As + (wm * 2 + 0) * 512 + lane * 16);
        v16h a1 = *(const v16h*)(As + (wm * 2 + 1) * 512 + lane * 16);
        v16h b0 = *(const v16h*)(Bs + (wn * 2 + 0) * 512 + lane * 16);
        v16h b1 = *(const v16h*)(Bs + (wn * 2 + 1) * 512 + lane * 16);
        acc00 = __builtin_amdgcn_wmma_f32_16x16x32_f16(false, a0, false, b0, (short)0, acc00, false, false);
        acc01 = __builtin_amdgcn_wmma_f32_16x16x32_f16(false, a0, false, b1, (short)0, acc01, false, false);
        acc10 = __builtin_amdgcn_wmma_f32_16x16x32_f16(false, a1, false, b0, (short)0, acc10, false, false);
        acc11 = __builtin_amdgcn_wmma_f32_16x16x32_f16(false, a1, false, b1, (short)0, acc11, false, false);
    }

    // ======================= tail: ragged K tile ===========================
    if (Kfull < K) {
        const int k0 = Kfull;
        float va0[8], va1[8];
#pragma unroll
        for (int i = 0; i < 8; ++i) {
            int pe = tid + (i << 8);
            int r  = pe >> 4;
            int k  = (pe & 15) << 1;
            int gr = row0 + r, gk = k0 + k;
            int cr  = gr < M ? gr : (M - 1);
            int ck0 = gk < K ? gk : (K - 1);
            int ck1 = (gk + 1) < K ? (gk + 1) : (K - 1);
            gcf base = (gcf)(A + (size_t)cr * K);
            va0[i] = base[ck0];
            va1[i] = base[ck1];
        }
        float vb0[4], vb1[4];
#pragma unroll
        for (int i = 0; i < 4; ++i) {
            int pe = tid + (i << 8);
            int kk = pe >> 5;
            int n  = (pe & 31) << 1;
            int gk = k0 + kk, gn = col0 + n;
            int ck  = gk < K ? gk : (K - 1);
            int cn0 = gn < N ? gn : (N - 1);
            int cn1 = (gn + 1) < N ? (gn + 1) : (N - 1);
            gcf base = (gcf)(W + (size_t)ck * N);
            vb0[i] = base[cn0];
            vb1[i] = base[cn1];
        }
        __syncthreads();
#pragma unroll
        for (int i = 0; i < 8; ++i) {
            int pe = tid + (i << 8);
            int k  = (pe & 15) << 1;
            int gk = k0 + k;
            union { _Float16 h[2]; unsigned u; } pk;
            pk.h[0] = (_Float16)((aOk[i] && gk < K)       ? va0[i] : 0.0f);
            pk.h[1] = (_Float16)((aOk[i] && (gk + 1) < K) ? va1[i] : 0.0f);
            *(unsigned*)aLds[i] = pk.u;
        }
#pragma unroll
        for (int i = 0; i < 4; ++i) {
            int pe = tid + (i << 8);
            int kk = pe >> 5;
            int gk = k0 + kk;
            union { _Float16 h[2]; unsigned u; } pk;
            pk.h[0] = (_Float16)((gk < K && bOk0[i]) ? vb0[i] : 0.0f);
            pk.h[1] = (_Float16)((gk < K && bOk1[i]) ? vb1[i] : 0.0f);
            bLds[i][0] = pk.h[0]; bLds[i][16] = pk.h[1];
        }
        __syncthreads();

        v16h a0 = *(const v16h*)(As + (wm * 2 + 0) * 512 + lane * 16);
        v16h a1 = *(const v16h*)(As + (wm * 2 + 1) * 512 + lane * 16);
        v16h b0 = *(const v16h*)(Bs + (wn * 2 + 0) * 512 + lane * 16);
        v16h b1 = *(const v16h*)(Bs + (wn * 2 + 1) * 512 + lane * 16);
        acc00 = __builtin_amdgcn_wmma_f32_16x16x32_f16(false, a0, false, b0, (short)0, acc00, false, false);
        acc01 = __builtin_amdgcn_wmma_f32_16x16x32_f16(false, a0, false, b1, (short)0, acc01, false, false);
        acc10 = __builtin_amdgcn_wmma_f32_16x16x32_f16(false, a1, false, b0, (short)0, acc10, false, false);
        acc11 = __builtin_amdgcn_wmma_f32_16x16x32_f16(false, a1, false, b1, (short)0, acc11, false, false);
    }

    // epilogue: C/D fragment layout (32-bit 16x16): lane 0-15 -> M=v, 16-31 -> M=v+8
    const int rb = row0 + wm * 32 + ((lane >> 4) << 3);
    const int cA = col0 + wn * 32 + (lane & 15);
    const int cB = cA + 16;
    const float biasA = (bias && cA < N) ? bias[cA] : 0.0f;
    const float biasB = (bias && cB < N) ? bias[cB] : 0.0f;
#pragma unroll
    for (int i = 0; i < 8; ++i) {
        int r0 = rb + i;        // sub-tile 0 (acc0x)
        int r1 = rb + 16 + i;   // sub-tile 1 (acc1x)
        if (r0 < M) {
            if (cA < N) { float v = acc00[i] + biasA; if (relu) v = fmaxf(v, 0.0f); C[(size_t)r0 * N + cA] = v; }
            if (cB < N) { float v = acc01[i] + biasB; if (relu) v = fmaxf(v, 0.0f); C[(size_t)r0 * N + cB] = v; }
        }
        if (r1 < M) {
            if (cA < N) { float v = acc10[i] + biasA; if (relu) v = fmaxf(v, 0.0f); C[(size_t)r1 * N + cA] = v; }
            if (cB < N) { float v = acc11[i] + biasB; if (relu) v = fmaxf(v, 0.0f); C[(size_t)r1 * N + cB] = v; }
        }
    }
}

// ---------------------------------------------------------------------------
// BatchNorm (training mode): per-channel mean/var over M rows, then apply.
// ---------------------------------------------------------------------------
__global__ __launch_bounds__(256)
void bn_stats(const float* __restrict__ x, int M, int C,
              float* __restrict__ mean, float* __restrict__ var)
{
    __shared__ float ss[256];
    __shared__ float sq[256];
    int c = blockIdx.x;
    float s = 0.0f, q = 0.0f;
    for (int r = threadIdx.x; r < M; r += 256) {
        float v = x[(size_t)r * C + c];
        s += v; q += v * v;
    }
    ss[threadIdx.x] = s; sq[threadIdx.x] = q;
    __syncthreads();
    for (int st = 128; st > 0; st >>= 1) {
        if ((int)threadIdx.x < st) {
            ss[threadIdx.x] += ss[threadIdx.x + st];
            sq[threadIdx.x] += sq[threadIdx.x + st];
        }
        __syncthreads();
    }
    if (threadIdx.x == 0) {
        float m = ss[0] / (float)M;
        mean[c] = m;
        var[c]  = sq[0] / (float)M - m * m;
    }
}

__global__ __launch_bounds__(256)
void bn_apply(const float* __restrict__ x, float* __restrict__ y, int total, int C,
              const float* __restrict__ mean, const float* __restrict__ var,
              const float* __restrict__ g, const float* __restrict__ b,
              const float* __restrict__ res, int relu)
{
    int i = blockIdx.x * 256 + threadIdx.x;
    if (i >= total) return;
    int c = i % C;
    float v = (x[i] - mean[c]) * rsqrtf(var[c] + 1e-5f) * g[c] + b[c];
    if (res)  v += res[i];
    if (relu) v = fmaxf(v, 0.0f);
    y[i] = v;
}

// ---------------------------------------------------------------------------
// Brute-force kNN: one thread per query, register-resident predicated top-K,
// reference points tiled through LDS. Writes indices (and squared dists).
// ---------------------------------------------------------------------------
template <int KN>
__global__ __launch_bounds__(256)
void knn_kernel(const float* __restrict__ q, int nq,
                const float* __restrict__ ref, int nref,
                int* __restrict__ idx, float* __restrict__ dist)
{
    __shared__ float sp[256 * 3];
    int qi = blockIdx.x * 256 + threadIdx.x;
    bool valid = qi < nq;
    int cqi = valid ? qi : (nq - 1);
    float qx = q[cqi * 3 + 0], qy = q[cqi * 3 + 1], qz = q[cqi * 3 + 2];

    float bd[KN]; int bi[KN];
#pragma unroll
    for (int j = 0; j < KN; ++j) { bd[j] = 1e30f; bi[j] = 0; }

    for (int t0 = 0; t0 < nref; t0 += 256) {
        int r  = t0 + threadIdx.x;
        int cr = r < nref ? r : (nref - 1);
        float rx = ref[cr * 3 + 0], ry = ref[cr * 3 + 1], rz = ref[cr * 3 + 2];
        bool ok = r < nref;
        sp[threadIdx.x * 3 + 0] = ok ? rx : 1e15f;
        sp[threadIdx.x * 3 + 1] = ok ? ry : 1e15f;
        sp[threadIdx.x * 3 + 2] = ok ? rz : 1e15f;
        __syncthreads();
        int cnt = (nref - t0) < 256 ? (nref - t0) : 256;
        for (int j = 0; j < cnt; ++j) {
            float dx = sp[j * 3 + 0] - qx;
            float dy = sp[j * 3 + 1] - qy;
            float dz = sp[j * 3 + 2] - qz;
            float d = dx * dx + dy * dy + dz * dz;
            int pos = 0;
#pragma unroll
            for (int u = 0; u < KN; ++u) pos += (bd[u] <= d) ? 1 : 0;
            if (pos < KN) {
#pragma unroll
                for (int u = KN - 1; u > 0; --u) {
                    bool mv = (u > pos);
                    bd[u] = mv ? bd[u - 1] : bd[u];
                    bi[u] = mv ? bi[u - 1] : bi[u];
                }
#pragma unroll
                for (int u = 0; u < KN; ++u) {
                    if (u == pos) { bd[u] = d; bi[u] = t0 + j; }
                }
            }
        }
        __syncthreads();
    }
    if (valid) {
#pragma unroll
        for (int j = 0; j < KN; ++j) {
            idx[(size_t)qi * KN + j] = bi[j];
            if (dist) dist[(size_t)qi * KN + j] = bd[j];
        }
    }
}

// ---------------------------------------------------------------------------
// Furthest point sampling: single workgroup (sequential dependency),
// register-resident min-distances + LDS argmax tree per iteration.
// ---------------------------------------------------------------------------
__global__ __launch_bounds__(1024)
void fps_kernel(const float* __restrict__ p, int N, int m, int* __restrict__ out)
{
    __shared__ float sd[1024];
    __shared__ int   si[1024];
    const int tid = threadIdx.x;
    float dist[16];
#pragma unroll
    for (int i = 0; i < 16; ++i) dist[i] = 1e30f;
    int last = 0;
    if (tid == 0) out[0] = 0;
    for (int it = 1; it < m; ++it) {
        float lx = p[last * 3 + 0], ly = p[last * 3 + 1], lz = p[last * 3 + 2];
        float bm = -1.0f; int bi = 0;
#pragma unroll
        for (int i = 0; i < 16; ++i) {
            int r = tid + (i << 10);
            if (r < N) {
                float dx = p[r * 3 + 0] - lx, dy = p[r * 3 + 1] - ly, dz = p[r * 3 + 2] - lz;
                float d = dx * dx + dy * dy + dz * dz;
                dist[i] = fminf(dist[i], d);
                if (dist[i] > bm) { bm = dist[i]; bi = r; }
            }
        }
        sd[tid] = bm; si[tid] = bi;
        __syncthreads();
        for (int s = 512; s > 0; s >>= 1) {
            if (tid < s) {
                if (sd[tid + s] > sd[tid]) { sd[tid] = sd[tid + s]; si[tid] = si[tid + s]; }
            }
            __syncthreads();
        }
        last = si[0];
        if (tid == 0) out[it] = last;
        __syncthreads();
    }
}

// ---------------------------------------------------------------------------
// Elementwise / gather helpers
// ---------------------------------------------------------------------------
__global__ __launch_bounds__(256)
void gather_rows(const float* __restrict__ src, const int* __restrict__ idx,
                 float* __restrict__ dst, int n, int C)
{
    int e = blockIdx.x * 256 + threadIdx.x;
    if (e >= n * C) return;
    dst[e] = src[(size_t)idx[e / C] * C + (e % C)];
}

__global__ __launch_bounds__(256)
void gather_nb(const float* __restrict__ src, const int* __restrict__ idx,
               float* __restrict__ dst, int n, int k, int C)
{
    int e = blockIdx.x * 256 + threadIdx.x;
    if (e >= n * k * C) return;
    int ch = e % C, ij = e / C;
    dst[e] = src[(size_t)idx[ij] * C + ch];
}

__global__ __launch_bounds__(256)
void rel_pos(const float* __restrict__ p, const int* __restrict__ idx,
             const float* __restrict__ ctr, float* __restrict__ rel, int n, int k)
{
    int e = blockIdx.x * 256 + threadIdx.x;
    if (e >= n * k * 3) return;
    int c = e % 3, ij = e / 3, i = ij / k;
    rel[e] = p[(size_t)idx[ij] * 3 + c] - ctr[(size_t)i * 3 + c];
}

__global__ __launch_bounds__(256)
void build_td_group(const float* __restrict__ p, const float* __restrict__ x,
                    const int* __restrict__ idx, const float* __restrict__ ctr,
                    float* __restrict__ g, int m, int k, int cin)
{
    int C = 3 + cin;
    int e = blockIdx.x * 256 + threadIdx.x;
    if (e >= m * k * C) return;
    int ch = e % C, ij = e / C, i = ij / k;
    int src = idx[ij];
    g[e] = (ch < 3) ? (p[(size_t)src * 3 + ch] - ctr[(size_t)i * 3 + ch])
                    : x[(size_t)src * cin + (ch - 3)];
}

__global__ __launch_bounds__(256)
void w_combine(const float* __restrict__ xkg, const float* __restrict__ xq,
               const float* __restrict__ pe, float* __restrict__ w, int n, int k, int c)
{
    int e = blockIdx.x * 256 + threadIdx.x;
    if (e >= n * k * c) return;
    int ch = e % c, i = (e / c) / k;
    w[e] = xkg[e] - xq[(size_t)i * c + ch] + pe[e];
}

__global__ __launch_bounds__(256)
void softmax_nb(float* __restrict__ w, int n, int k, int cs)
{
    int e = blockIdx.x * 256 + threadIdx.x;
    if (e >= n * cs) return;
    int i = e / cs, t = e % cs;
    size_t base = (size_t)i * k * cs + t;
    float mx = -1e30f;
    for (int j = 0; j < k; ++j) mx = fmaxf(mx, w[base + (size_t)j * cs]);
    float s = 0.0f;
    for (int j = 0; j < k; ++j) {
        float v = expf(w[base + (size_t)j * cs] - mx);
        w[base + (size_t)j * cs] = v;
        s += v;
    }
    float inv = 1.0f / s;
    for (int j = 0; j < k; ++j) w[base + (size_t)j * cs] *= inv;
}

__global__ __launch_bounds__(256)
void pt_aggregate(const float* __restrict__ xv, const float* __restrict__ pe,
                  const float* __restrict__ w, float* __restrict__ out,
                  int n, int k, int c, int cs)
{
    int e = blockIdx.x * 256 + threadIdx.x;
    if (e >= n * c) return;
    int i = e / c, ch = e % c, t = ch % cs;
    float s = 0.0f;
    for (int j = 0; j < k; ++j) {
        size_t o = (size_t)i * k + j;
        s += (xv[o * c + ch] + pe[o * c + ch]) * w[o * cs + t];
    }
    out[e] = s;
}

__global__ __launch_bounds__(256)
void rowmax_nb(const float* __restrict__ g, float* __restrict__ out, int m, int k, int c)
{
    int e = blockIdx.x * 256 + threadIdx.x;
    if (e >= m * c) return;
    int i = e / c, ch = e % c;
    float mx = -1e30f;
    for (int j = 0; j < k; ++j) mx = fmaxf(mx, g[((size_t)i * k + j) * c + ch]);
    out[e] = mx;
}

__global__ __launch_bounds__(256)
void concat2(const float* __restrict__ a, int Ca, const float* __restrict__ b, int Cb,
             float* __restrict__ out, int M, int bbc)
{
    int C = Ca + Cb;
    int e = blockIdx.x * 256 + threadIdx.x;
    if (e >= M * C) return;
    int i = e / C, ch = e % C;
    out[e] = (ch < Ca) ? a[(size_t)i * Ca + ch]
                       : b[(size_t)(bbc ? 0 : i) * Cb + (ch - Ca)];
}

__global__ __launch_bounds__(256)
void interp_add(const float* __restrict__ a, const float* __restrict__ xc,
                const int* __restrict__ idx, const float* __restrict__ d2,
                float* __restrict__ out, int M, int C)
{
    int e = blockIdx.x * 256 + threadIdx.x;
    if (e >= M * C) return;
    int i = e / C, ch = e % C;
    float s = 0.0f, ws = 0.0f;
#pragma unroll
    for (int j = 0; j < 3; ++j) {
        float d = sqrtf(d2[(size_t)i * 3 + j] + 1e-12f);
        float w = 1.0f / (d + 1e-8f);
        ws += w;
        s += w * xc[(size_t)idx[(size_t)i * 3 + j] * C + ch];
    }
    out[e] = a[e] + s / ws;
}

// ===========================================================================
// Host orchestration
// ===========================================================================
namespace {
struct Lin { const float* W; const float* b; int fin, fout; };
struct BN  { const float* b; const float* g; };
struct Blk {
    BN bn1, bn2, bn3, bnp, bnw1, bnw2;
    Lin k, lin1, lin3, p1, p2, q, v, w1, w2;
};
struct DecTU { BN bn1, bn2; Lin lin1, lin2; };
} // namespace

extern "C" void kernel_launch(void* const* d_in, const int* in_sizes, int n_in,
                              void* d_out, int out_size, void* d_ws, size_t ws_size,
                              hipStream_t stream)
{
    (void)in_sizes; (void)n_in; (void)out_size;

    const float* p0 = (const float*)d_in[0];
    const float* x0 = (const float*)d_in[1];
    // d_in[2] = o0 (unused: single cloud)

    // -------- parameter pytree traversal (alphabetical keys, jax order) ----
    int cur = 3;
    auto nf = [&]() -> const float* { return (const float*)d_in[cur++]; };
    auto rdBN  = [&](BN& bn) { bn.b = nf(); bn.g = nf(); };
    auto rdLin = [&](Lin& L, int fi, int fo, bool bias) {
        L.W = nf(); L.b = bias ? nf() : nullptr; L.fin = fi; L.fout = fo;
    };
    auto rdBlk = [&](Blk& B, int c) {
        int cs = c / 8;
        rdBN(B.bn1); rdBN(B.bn2); rdBN(B.bn3); rdBN(B.bnp); rdBN(B.bnw1); rdBN(B.bnw2);
        rdLin(B.k, c, c, true);
        rdLin(B.lin1, c, c, false);
        rdLin(B.lin3, c, c, false);
        rdLin(B.p1, 3, 3, true);
        rdLin(B.p2, 3, c, true);
        rdLin(B.q, c, c, true);
        rdLin(B.v, c, c, true);
        rdLin(B.w1, c, cs, true);
        rdLin(B.w2, cs, cs, true);
    };

    static const int NPTS[5] = {16384, 4096, 1024, 256, 64};
    static const int CH[5]   = {32, 64, 128, 256, 512};
    static const int NBLK[5] = {1, 2, 3, 5, 2};          // NUM_BLOCK - 1
    static const int FIN[5]  = {6, 35, 67, 131, 259};    // td input channels

    // top-level sorted: cls, dec1..dec5, enc1..enc5
    BN cls_bn; Lin cls_l1, cls_l2;
    rdBN(cls_bn); rdLin(cls_l1, 32, 32, true); rdLin(cls_l2, 32, 13, true);

    Blk decBlk[6]; DecTU decTU[5]; BN tu5_bn1; Lin tu5_lin1, tu5_lin2;
    for (int l = 1; l <= 4; ++l) {                       // dec1..dec4
        int cout = CH[l - 1], cin = CH[l];
        rdBlk(decBlk[l], cout);                          // "blocks" < "tu"
        rdBN(decTU[l].bn1); rdBN(decTU[l].bn2);
        rdLin(decTU[l].lin1, cout, cout, true);
        rdLin(decTU[l].lin2, cin, cout, true);
    }
    rdBlk(decBlk[5], 512);                               // dec5
    rdBN(tu5_bn1); rdLin(tu5_lin1, 1024, 512, true); rdLin(tu5_lin2, 512, 512, true);

    Blk encBlk[5][5]; BN encTDbn[5]; Lin encTDlin[5];
    for (int i = 0; i < 5; ++i) {                        // enc1..enc5
        for (int j = 0; j < NBLK[i]; ++j) rdBlk(encBlk[i][j], CH[i]);
        rdBN(encTDbn[i]); rdLin(encTDlin[i], FIN[i], CH[i], false);
    }

    // -------- workspace arena: front = persistent, back = scratch ----------
    char* wsBase = (char*)d_ws;
    size_t wsCap = ws_size & ~(size_t)255;
    size_t frontOff = 0, backOff = 0;
    auto fAlloc = [&](size_t bytes) -> void* {
        void* p = wsBase + frontOff;
        frontOff += (bytes + 255) & ~(size_t)255;
        return p;
    };
    auto sAlloc = [&](size_t bytes) -> void* {
        backOff += (bytes + 255) & ~(size_t)255;
        return wsBase + wsCap - backOff;
    };
    auto sReset = [&]() { backOff = 0; };
    auto ff  = [&](size_t n) -> float* { return (float*)fAlloc(n * 4); };
    auto fi_ = [&](size_t n) -> int*   { return (int*)fAlloc(n * 4); };
    auto sfp = [&](size_t n) -> float* { return (float*)sAlloc(n * 4); };
    auto sip = [&](size_t n) -> int*   { return (int*)sAlloc(n * 4); };

    auto blocks1d = [](int total) { return dim3((unsigned)((total + 255) / 256)); };

    // -------- op launchers -------------------------------------------------
    auto gemm = [&](const float* A, const Lin& L, float* out, int M, int relu) {
        dim3 grid((unsigned)((L.fout + 63) / 64), (unsigned)((M + 127) / 128));
        gemm_f16wmma<<<grid, dim3(256), 0, stream>>>(A, L.W, L.b, out, M, L.fin, L.fout, relu);
    };
    auto bn_full = [&](const float* xin, float* yout, int M, int C, const BN& bn,
                       const float* res, int relu) {
        float* mv = sfp((size_t)2 * C);
        bn_stats<<<dim3((unsigned)C), dim3(256), 0, stream>>>(xin, M, C, mv, mv + C);
        int total = M * C;
        bn_apply<<<blocks1d(total), dim3(256), 0, stream>>>(
            xin, yout, total, C, mv, mv + C, bn.g, bn.b, res, relu);
    };

    // -------- point transformer block --------------------------------------
    auto run_block = [&](const Blk& B, const float* p, float* x, const int* nn, int n, int c) {
        sReset();
        int cs = c / 8, nk = n * 16;
        float* y = sfp((size_t)n * c);
        gemm(x, B.lin1, y, n, 0);
        bn_full(y, y, n, c, B.bn1, nullptr, 1);

        float* xq = sfp((size_t)n * c); gemm(y, B.q, xq, n, 0);
        float* xk = sfp((size_t)n * c); gemm(y, B.k, xk, n, 0);
        float* xv = sfp((size_t)n * c); gemm(y, B.v, xv, n, 0);

        float* rel = sfp((size_t)nk * 3);
        rel_pos<<<blocks1d(nk * 3), dim3(256), 0, stream>>>(p, nn, p, rel, n, 16);
        float* t1 = sfp((size_t)nk * 3);
        gemm(rel, B.p1, t1, nk, 0);
        bn_full(t1, t1, nk, 3, B.bnp, nullptr, 1);
        float* pe = sfp((size_t)nk * c);
        gemm(t1, B.p2, pe, nk, 0);

        float* xkg = sfp((size_t)nk * c);
        gather_nb<<<blocks1d(nk * c), dim3(256), 0, stream>>>(xk, nn, xkg, n, 16, c);
        float* xvg = sfp((size_t)nk * c);
        gather_nb<<<blocks1d(nk * c), dim3(256), 0, stream>>>(xv, nn, xvg, n, 16, c);

        float* w = sfp((size_t)nk * c);
        w_combine<<<blocks1d(nk * c), dim3(256), 0, stream>>>(xkg, xq, pe, w, n, 16, c);
        bn_full(w, w, nk, c, B.bnw1, nullptr, 1);
        float* w1o = sfp((size_t)nk * cs);
        gemm(w, B.w1, w1o, nk, 0);
        bn_full(w1o, w1o, nk, cs, B.bnw2, nullptr, 1);
        float* w2o = sfp((size_t)nk * cs);
        gemm(w1o, B.w2, w2o, nk, 0);
        softmax_nb<<<blocks1d(n * cs), dim3(256), 0, stream>>>(w2o, n, 16, cs);

        float* agg = sfp((size_t)n * c);
        pt_aggregate<<<blocks1d(n * c), dim3(256), 0, stream>>>(xvg, pe, w2o, agg, n, 16, c, cs);
        bn_full(agg, agg, n, c, B.bn2, nullptr, 1);
        float* y3 = sfp((size_t)n * c);
        gemm(agg, B.lin3, y3, n, 0);
        bn_full(y3, x, n, c, B.bn3, x, 1);   // x = relu(bn3(y3) + x)
    };

    // ======================= encoder =======================================
    const float* ps[5];
    float* xs[5];
    int* knnS[5];

    // enc1 (stride 1): x = relu(bn(linear(concat(p0, x0))))
    sReset();
    float* x0c = sfp((size_t)16384 * 6);
    concat2<<<blocks1d(16384 * 6), dim3(256), 0, stream>>>(p0, 3, x0, 3, x0c, 16384, 0);
    ps[0] = p0;
    xs[0] = ff((size_t)16384 * 32);
    gemm(x0c, encTDlin[0], xs[0], 16384, 0);
    bn_full(xs[0], xs[0], 16384, 32, encTDbn[0], nullptr, 1);
    knnS[0] = fi_((size_t)16384 * 16);
    knn_kernel<16><<<blocks1d(16384), dim3(256), 0, stream>>>(ps[0], 16384, ps[0], 16384, knnS[0], nullptr);
    for (int j = 0; j < NBLK[0]; ++j) run_block(encBlk[0][j], ps[0], xs[0], knnS[0], 16384, 32);

    for (int i = 1; i < 5; ++i) {
        int npv = NPTS[i - 1], n = NPTS[i], cin = CH[i - 1], c = CH[i];
        sReset();
        int* sidx = sip((size_t)n);
        fps_kernel<<<dim3(1), dim3(1024), 0, stream>>>(ps[i - 1], npv, n, sidx);
        float* pnew = ff((size_t)n * 3);
        gather_rows<<<blocks1d(n * 3), dim3(256), 0, stream>>>(ps[i - 1], sidx, pnew, n, 3);
        ps[i] = pnew;
        int* tdidx = sip((size_t)n * 16);
        knn_kernel<16><<<blocks1d(n), dim3(256), 0, stream>>>(pnew, n, ps[i - 1], npv, tdidx, nullptr);
        float* gbuf = sfp((size_t)n * 16 * (3 + cin));
        build_td_group<<<blocks1d(n * 16 * (3 + cin)), dim3(256), 0, stream>>>(
            ps[i - 1], xs[i - 1], tdidx, pnew, gbuf, n, 16, cin);
        float* lo = sfp((size_t)n * 16 * c);
        gemm(gbuf, encTDlin[i], lo, n * 16, 0);
        bn_full(lo, lo, n * 16, c, encTDbn[i], nullptr, 1);
        xs[i] = ff((size_t)n * c);
        rowmax_nb<<<blocks1d(n * c), dim3(256), 0, stream>>>(lo, xs[i], n, 16, c);
        knnS[i] = fi_((size_t)n * 16);
        knn_kernel<16><<<blocks1d(n), dim3(256), 0, stream>>>(pnew, n, pnew, n, knnS[i], nullptr);
        for (int j = 0; j < NBLK[i]; ++j) run_block(encBlk[i][j], ps[i], xs[i], knnS[i], n, c);
    }

    // ======================= decoder =======================================
    // dec5: transition_up_head + 1 block at (64, 512)
    sReset();
    float* mv5 = sfp((size_t)2 * 512);
    bn_stats<<<dim3(512), dim3(256), 0, stream>>>(xs[4], 64, 512, mv5, mv5 + 512); // mean
    float* gvec = sfp((size_t)512);
    gemm(mv5, tu5_lin2, gvec, 1, 1);                     // relu(linear(mean))
    float* xg = sfp((size_t)64 * 1024);
    concat2<<<blocks1d(64 * 1024), dim3(256), 0, stream>>>(xs[4], 512, gvec, 512, xg, 64, 1);
    float* x5 = ff((size_t)64 * 512);
    gemm(xg, tu5_lin1, x5, 64, 0);
    bn_full(x5, x5, 64, 512, tu5_bn1, nullptr, 1);
    run_block(decBlk[5], ps[4], x5, knnS[4], 64, 512);

    float* xcur = x5;
    int ncur = 64;
    for (int l = 4; l >= 1; --l) {                       // dec4..dec1 (levels 3..0)
        int nfp = NPTS[l - 1], cf = CH[l - 1];
        sReset();
        float* a = sfp((size_t)nfp * cf);
        gemm(xs[l - 1], decTU[l].lin1, a, nfp, 0);
        bn_full(a, a, nfp, cf, decTU[l].bn1, nullptr, 1);
        float* bb = sfp((size_t)ncur * cf);
        gemm(xcur, decTU[l].lin2, bb, ncur, 0);
        bn_full(bb, bb, ncur, cf, decTU[l].bn2, nullptr, 1);
        int* idx3 = sip((size_t)nfp * 3);
        float* d3 = sfp((size_t)nfp * 3);
        knn_kernel<3><<<blocks1d(nfp), dim3(256), 0, stream>>>(ps[l - 1], nfp, ps[l], ncur, idx3, d3);
        float* xnew = ff((size_t)nfp * cf);
        interp_add<<<blocks1d(nfp * cf), dim3(256), 0, stream>>>(a, bb, idx3, d3, xnew, nfp, cf);
        run_block(decBlk[l], ps[l - 1], xnew, knnS[l - 1], nfp, cf);
        xcur = xnew;
        ncur = nfp;
    }

    // ======================= classifier head ===============================
    sReset();
    float* yh = sfp((size_t)16384 * 32);
    gemm(xcur, cls_l1, yh, 16384, 0);
    bn_full(yh, yh, 16384, 32, cls_bn, nullptr, 1);
    gemm(yh, cls_l2, (float*)d_out, 16384, 0);           // [16384, 13] f32
}
